// Multi_Gate_MOE_64424509440700
// MI455X (gfx1250) — matrix operand order repaired and checked by reading, not compiled, not run on hardware
//
#include <hip/hip_runtime.h>
#include <hip/hip_bf16.h>

#define B_   16384
#define IN_  1024
#define H_   512
#define D_   256
#define E_   8
#define T_   4
#define BM   64
#define XP   1032   // x LDS pitch (bf16 elems): row stride 2064B -> conflict-free b128 reads
#define HP   520    // h LDS pitch (bf16 elems): row stride 1040B -> conflict-free b128 reads

typedef __attribute__((ext_vector_type(16))) __bf16        v16bf;
typedef __attribute__((ext_vector_type(8)))  float         v8f;
typedef __attribute__((ext_vector_type(4)))  unsigned int  v4u;
typedef __attribute__((ext_vector_type(8)))  unsigned int  v8u;

static __device__ __forceinline__ v16bf ab16(v4u lo, v4u hi) {
    union { v4u q[2]; v16bf v; } c; c.q[0] = lo; c.q[1] = hi; return c.v;
}
static __device__ __forceinline__ v16bf ab32(v8u x) {
    union { v8u q; v16bf v; } c; c.q = x; return c.v;
}
static __device__ __forceinline__ unsigned short f2bf(float f) {
    unsigned int u = __float_as_uint(f);
    unsigned int r = u + 0x7FFFu + ((u >> 16) & 1u);   // round-to-nearest-even
    return (unsigned short)(r >> 16);
}

// ---------------- f32 -> bf16 convert ----------------
__global__ void cvt_kernel(const float* __restrict__ in, unsigned short* __restrict__ out, int n) {
    int i = blockIdx.x * blockDim.x + threadIdx.x;
    int stride = gridDim.x * blockDim.x;
    for (; i < n; i += stride) out[i] = f2bf(in[i]);
}

// ---------------- gate softmax: w[b][t*8+e] ----------------
// one wave32 per batch row; lane = t*8+e
__global__ void __launch_bounds__(256) gate_kernel(const float* __restrict__ x,
                                                   const float* __restrict__ Wg,
                                                   const float* __restrict__ bg,
                                                   float* __restrict__ gw) {
    int wave = threadIdx.x >> 5;
    int lane = threadIdx.x & 31;
    int b = blockIdx.x * 8 + wave;
    const float4* xr = (const float4*)(x + (size_t)b * IN_);
    const float4* gr = (const float4*)(Wg + (size_t)lane * IN_);  // Wg flat [T*E][IN]
    float acc = 0.f;
    for (int k = 0; k < IN_ / 4; ++k) {
        float4 xv = xr[k]; float4 gv = gr[k];
        acc += xv.x * gv.x + xv.y * gv.y + xv.z * gv.z + xv.w * gv.w;
    }
    acc += bg[lane];
    float m = acc;
    m = fmaxf(m, __shfl_xor(m, 1, 8));
    m = fmaxf(m, __shfl_xor(m, 2, 8));
    m = fmaxf(m, __shfl_xor(m, 4, 8));
    float ev = __expf(acc - m);
    float s = ev;
    s += __shfl_xor(s, 1, 8);
    s += __shfl_xor(s, 2, 8);
    s += __shfl_xor(s, 4, 8);
    gw[(size_t)b * 32 + lane] = ev / s;
}

// ---------------- fused MoE: both GEMMs + combine, per 64-row tile ----------------
// 16 waves: 2 row-groups (32 rows) x 8 col-groups. Each wave: 2 A-tiles per B-tile
// so every (uncoalesced) weight load feeds two v_wmma ops; B loads are software-
// pipelined one tile ahead of the WMMAs that consume them.
__global__ void __launch_bounds__(512) moe_kernel(const unsigned short* __restrict__ xb,
                                                  const unsigned short* __restrict__ W1b,
                                                  const unsigned short* __restrict__ W2b,
                                                  const float* __restrict__ b1,
                                                  const float* __restrict__ b2,
                                                  const float* __restrict__ gw,
                                                  float* __restrict__ out) {
    extern __shared__ char smem[];
    unsigned short* sx = (unsigned short*)smem;                          // [BM][XP]
    unsigned short* sh = (unsigned short*)(smem + BM * XP * 2);          // [BM][HP]
    float*          sw = (float*)(smem + BM * XP * 2 + BM * HP * 2);     // [BM][32]

    const int tid = threadIdx.x;
    const int b0  = blockIdx.x * BM;

    // stage x tile (bf16, pitch-padded) and gate weights into LDS (16B-vectorized)
    {
        const uint4* src = (const uint4*)xb + (size_t)b0 * (IN_ / 8);
        uint4* dst = (uint4*)sx;                  // XP/8 = 129 uint4 per row
        for (int i = tid; i < BM * (IN_ / 8); i += 512) {
            int row = i >> 7, cp = i & 127;
            dst[row * (XP / 8) + cp] = src[row * (IN_ / 8) + cp];
        }
        for (int i = tid; i < BM * 32; i += 512) sw[i] = gw[(size_t)b0 * 32 + i];
    }
    __syncthreads();

    const int wave = tid >> 5;
    const int lane = tid & 31;
    const int half = lane >> 4;
    const int ln   = lane & 15;
    const int r    = wave >> 3;     // row group: rows [r*32, r*32+32)
    const int c    = wave & 7;      // col group
    const int arow0 = r * 32 + ln;
    const int arow1 = r * 32 + 16 + ln;

    float oacc[T_][2][2][8] = {};   // [task][row tile][d tile][C vgpr]

    for (int e = 0; e < E_; ++e) {
        // ---- phase 1: h = relu(x @ W1_e^T + b1_e); this wave: h-cols [c*64, c*64+64)
        v8f hacc[2][4] = {};
        const unsigned short* W1e = W1b + (size_t)e * H_ * IN_;
        for (int k0 = 0; k0 < IN_; k0 += 32) {
            const unsigned short* ap0 = sx + arow0 * XP + k0 + half * 8;
            const unsigned short* ap1 = sx + arow1 * XP + k0 + half * 8;
            v16bf A0 = ab16(*(const v4u*)ap0, *(const v4u*)(ap0 + 16));
            v16bf A1 = ab16(*(const v4u*)ap1, *(const v4u*)(ap1 + 16));
            const unsigned short* bp0 = W1e + (size_t)(c * 64 + ln) * IN_ + k0 + half * 8;
            v16bf bcur = ab16(*(const v4u*)bp0, *(const v4u*)(bp0 + 16));
            #pragma unroll
            for (int j = 0; j < 4; ++j) {
                v16bf bnext = bcur;
                if (j < 3) {    // prefetch next B tile before consuming current
                    const unsigned short* bpn = W1e + (size_t)(c * 64 + (j + 1) * 16 + ln) * IN_ +
                                          k0 + half * 8;
                    bnext = ab16(*(const v4u*)bpn, *(const v4u*)(bpn + 16));
                }
                v16bf Bv = bcur;
                hacc[0][j] = __builtin_amdgcn_wmma_f32_16x16x32_bf16(
                    false, A0, false, Bv, (short)0, hacc[0][j], false, false);
                hacc[1][j] = __builtin_amdgcn_wmma_f32_16x16x32_bf16(
                    false, A1, false, Bv, (short)0, hacc[1][j], false, false);
                bcur = bnext;
            }
        }
        #pragma unroll
        for (int j = 0; j < 4; ++j) {
            int n = c * 64 + j * 16 + ln;
            float bias = b1[e * H_ + n];
            #pragma unroll
            for (int rt = 0; rt < 2; ++rt) {
                #pragma unroll
                for (int i = 0; i < 8; ++i) {
                    float v = hacc[rt][j][i] + bias;
                    v = v > 0.f ? v : 0.f;
                    sh[(r * 32 + rt * 16 + i + 8 * half) * HP + n] = f2bf(v);
                }
            }
        }
        __syncthreads();

        // ---- phase 2: feats = relu(h @ W2_e^T + b2_e); this wave: d-cols [c*32, c*32+32)
        v8f facc[2][2] = {};
        const unsigned short* W2e = W2b + (size_t)e * D_ * H_;
        for (int k0 = 0; k0 < H_; k0 += 32) {
            const unsigned short* ap0 = sh + arow0 * HP + k0 + half * 8;
            const unsigned short* ap1 = sh + arow1 * HP + k0 + half * 8;
            v16bf A0 = ab16(*(const v4u*)ap0, *(const v4u*)(ap0 + 16));
            v16bf A1 = ab16(*(const v4u*)ap1, *(const v4u*)(ap1 + 16));
            const unsigned short* bq0 = W2e + (size_t)(c * 32 + ln) * H_ + k0 + half * 8;
            v16bf bcur = ab16(*(const v4u*)bq0, *(const v4u*)(bq0 + 16));
            #pragma unroll
            for (int j = 0; j < 2; ++j) {
                v16bf bnext = bcur;
                if (j < 1) {
                    const unsigned short* bqn = W2e + (size_t)(c * 32 + 16 + ln) * H_ +
                                          k0 + half * 8;
                    bnext = ab16(*(const v4u*)bqn, *(const v4u*)(bqn + 16));
                }
                v16bf Bv = bcur;
                facc[0][j] = __builtin_amdgcn_wmma_f32_16x16x32_bf16(
                    false, A0, false, Bv, (short)0, facc[0][j], false, false);
                facc[1][j] = __builtin_amdgcn_wmma_f32_16x16x32_bf16(
                    false, A1, false, Bv, (short)0, facc[1][j], false, false);
                bcur = bnext;
            }
        }
        __syncthreads();    // all h reads done before next expert overwrites sh

        #pragma unroll
        for (int j = 0; j < 2; ++j) {
            int n = c * 32 + j * 16 + ln;
            float bias = b2[e * D_ + n];
            #pragma unroll
            for (int rt = 0; rt < 2; ++rt) {
                #pragma unroll
                for (int i = 0; i < 8; ++i) {
                    float v = facc[rt][j][i] + bias;
                    facc[rt][j][i] = v > 0.f ? v : 0.f;
                }
            }
        }
        // weighted combine into register accumulators: out[t] += w[b,t,e] * feats
        #pragma unroll
        for (int t = 0; t < T_; ++t) {
            #pragma unroll
            for (int rt = 0; rt < 2; ++rt) {
                #pragma unroll
                for (int i = 0; i < 8; ++i) {
                    float wv = sw[(r * 32 + rt * 16 + i + 8 * half) * 32 + t * 8 + e];
                    #pragma unroll
                    for (int j = 0; j < 2; ++j)
                        oacc[t][rt][j][i] += wv * facc[rt][j][i];
                }
            }
        }
    }

    // epilogue: out[t][b][d]
    #pragma unroll
    for (int t = 0; t < T_; ++t) {
        #pragma unroll
        for (int rt = 0; rt < 2; ++rt) {
            #pragma unroll
            for (int j = 0; j < 2; ++j) {
                int d = c * 32 + j * 16 + ln;
                #pragma unroll
                for (int i = 0; i < 8; ++i) {
                    int b = b0 + r * 32 + rt * 16 + i + 8 * half;
                    out[((size_t)t * B_ + b) * D_ + d] = oacc[t][rt][j][i];
                }
            }
        }
    }
}

extern "C" void kernel_launch(void* const* d_in, const int* in_sizes, int n_in,
                              void* d_out, int out_size, void* d_ws, size_t ws_size,
                              hipStream_t stream) {
    const float* x  = (const float*)d_in[0];
    const float* W1 = (const float*)d_in[1];
    const float* b1 = (const float*)d_in[2];
    const float* W2 = (const float*)d_in[3];
    const float* b2 = (const float*)d_in[4];
    const float* Wg = (const float*)d_in[5];
    const float* bg = (const float*)d_in[6];
    float* out = (float*)d_out;

    char* ws = (char*)d_ws;
    unsigned short* xb  = (unsigned short*)ws;                         // 32 MB
    unsigned short* W1b = (unsigned short*)(ws + 33554432);            // 8 MB
    unsigned short* W2b = (unsigned short*)(ws + 41943040);            // 2 MB
    float*          gwp = (float*)(ws + 44040192);                     // 2 MB

    size_t smem = (size_t)BM * XP * 2 + (size_t)BM * HP * 2 + (size_t)BM * 32 * 4;
    (void)hipFuncSetAttribute((const void*)moe_kernel,
                              hipFuncAttributeMaxDynamicSharedMemorySize, (int)smem);

    cvt_kernel<<<4096, 256, 0, stream>>>(x,  xb,  B_ * IN_);
    cvt_kernel<<<2048, 256, 0, stream>>>(W1, W1b, E_ * H_ * IN_);
    cvt_kernel<<<1024, 256, 0, stream>>>(W2, W2b, E_ * D_ * H_);
    gate_kernel<<<B_ / 8, 256, 0, stream>>>(x, Wg, bg, gwp);
    moe_kernel<<<B_ / BM, 512, smem, stream>>>(xb, W1b, W2b, b1, b2, gwp, out);
}
